// MACE_model_56495999812200
// MI455X (gfx1250) — matrix operand order repaired and checked by reading, not compiled, not run on hardware
//
#include <hip/hip_runtime.h>

typedef __attribute__((ext_vector_type(16))) _Float16 v16h;
typedef __attribute__((ext_vector_type(8)))  _Float16 v8h;
typedef __attribute__((ext_vector_type(8)))  float    v8f;

#define NNODE 10000
#define NEDGE 160000
#define KCH   128
#define NGR   64

__device__ __forceinline__ float siluf(float x) {
    float s = 1.f / (1.f + __expf(-x));
    return x * s;
}
__device__ __forceinline__ float dsiluf(float x) {
    float s = 1.f / (1.f + __expf(-x));
    return s * (1.f + x * (1.f - s));
}

// ---------------------------------------------------------------------------
// Weight prep: 7 f32 128x128 weights -> f16 [0..6] + 6 transposes [7..12],
// stored PRE-SWIZZLED in WMMA B-fragment order:
//   halfIndex = ct*2048 + kt*512 + lane*16 + i
//   where ct=col>>4, kt=row>>5, lane=(col&15)+16*((row>>4)&1), i=row&15
// so a wave's B fragment is one contiguous 32-byte v16h load per lane.
// 7<-2 (W_h1h1p^T) 8<-3 (WA0^T) 9<-4 (WA1^T) 10<-5 (W_h1h2^T) 11<-6 (W_m2h2^T) 12<-1 (W_m1h1^T)
// ---------------------------------------------------------------------------
__global__ __launch_bounds__(256) void prep_weights(
    const float* __restrict__ w0, const float* __restrict__ w1,
    const float* __restrict__ w2, const float* __restrict__ w3,
    const float* __restrict__ w4, const float* __restrict__ w5,
    const float* __restrict__ w6, _Float16* __restrict__ dst)
{
    int tid = blockIdx.x * 256 + threadIdx.x;
    if (tid >= 13 * 128 * 128) return;
    int mat = tid >> 14;
    int pos = tid & 16383;
    int r = pos >> 7, c = pos & 127;      // r = K index, c = N column
    int sm = (mat < 7) ? mat : ((mat == 12) ? 1 : (mat - 5));
    const float* S;
    switch (sm) {
        case 0: S = w0; break; case 1: S = w1; break; case 2: S = w2; break;
        case 3: S = w3; break; case 4: S = w4; break; case 5: S = w5; break;
        default: S = w6; break;
    }
    float v = (mat >= 7) ? S[c * 128 + r] : S[r * 128 + c];
    int lane = (c & 15) + (((r >> 4) & 1) << 4);
    int dsti = ((c >> 4) << 11) + ((r >> 5) << 9) + (lane << 4) + (r & 15);
    dst[(long long)mat * 16384 + dsti] = (_Float16)v;
}

// ---------------------------------------------------------------------------
// h0 = Wh0[species] + bh0
// ---------------------------------------------------------------------------
__global__ __launch_bounds__(256) void node_h0_k(
    const float* __restrict__ Wh0, const float* __restrict__ bh0,
    const int* __restrict__ spec, float* __restrict__ h0)
{
    int idx = blockIdx.x * 256 + threadIdx.x;
    if (idx >= NNODE * KCH) return;
    int n = idx >> 7, k = idx & 127;
    h0[idx] = Wh0[spec[n] * KCH + k] + bh0[k];
}

// ---------------------------------------------------------------------------
// WMMA GEMM: C[M x 128] = (bias|Cin|0) + A0@B0 (+ A1@B1)
// A f32 row-major [M x 128]; B f16 pre-swizzled fragments; C f32 row-major.
// Block = 256 (8 waves); wave w -> 16-col tile w; blockIdx.x -> 16-row tile.
// A tile staged once per block in LDS as f16 (cooperative), read via ds_load.
// ---------------------------------------------------------------------------
__global__ __launch_bounds__(256) void gemm_nk(
    const float* __restrict__ A0, const _Float16* __restrict__ B0,
    const float* __restrict__ A1, const _Float16* __restrict__ B1,
    const float* __restrict__ bias, float* __restrict__ C,
    int M, int accum)
{
    __shared__ __align__(32) _Float16 Atile[16 * KCH];   // 4 KB
    const int wave  = threadIdx.x >> 5;
    const int lane  = threadIdx.x & 31;
    const int row0  = blockIdx.x * 16;
    if (row0 >= M) return;                // uniform over block
    const int col16 = lane & 15;
    const int hi    = lane >> 4;
    const int col   = wave * 16 + col16;

    union { float f[8]; v8f v; } acc;
    if (accum) {
        #pragma unroll
        for (int i = 0; i < 8; i++) {
            int m = row0 + i + 8 * hi; if (m >= M) m = M - 1;
            acc.f[i] = C[(long long)m * KCH + col];
        }
    } else if (bias) {
        float b = bias[col];
        #pragma unroll
        for (int i = 0; i < 8; i++) acc.f[i] = b;
    } else {
        #pragma unroll
        for (int i = 0; i < 8; i++) acc.f[i] = 0.f;
    }

    const int srow = threadIdx.x >> 4;          // 0..15
    const int scol = (threadIdx.x & 15) * 8;    // 0..120

    for (int pass = 0; pass < 2; ++pass) {
        const float*    A = pass ? A1 : A0;
        const _Float16* B = pass ? B1 : B0;
        if (A == nullptr) continue;             // uniform: EXEC stays full
        __syncthreads();
        {   // cooperative stage: 16x128 f32 -> f16 in LDS
            int m = row0 + srow; if (m >= M) m = M - 1;
            const float4 lo = *(const float4*)(A + (long long)m * KCH + scol);
            const float4 hv = *(const float4*)(A + (long long)m * KCH + scol + 4);
            union { _Float16 h[8]; v8h v; } p;
            p.h[0] = (_Float16)lo.x; p.h[1] = (_Float16)lo.y;
            p.h[2] = (_Float16)lo.z; p.h[3] = (_Float16)lo.w;
            p.h[4] = (_Float16)hv.x; p.h[5] = (_Float16)hv.y;
            p.h[6] = (_Float16)hv.z; p.h[7] = (_Float16)hv.w;
            *(v8h*)&Atile[srow * KCH + scol] = p.v;
        }
        __syncthreads();
        #pragma unroll
        for (int kt = 0; kt < 4; kt++) {
            // A 16x32 fragment: lane m=col16; K = kt*32 + hi*16 + (0..15)
            int ka = kt * 32 + hi * 16;
            union { v8h h2[2]; v16h v; } af;
            af.h2[0] = *(const v8h*)&Atile[col16 * KCH + ka];
            af.h2[1] = *(const v8h*)&Atile[col16 * KCH + ka + 8];
            // B fragment: contiguous 16 halves per lane (pre-swizzled)
            v16h bf = *(const v16h*)(B + (wave << 11) + (kt << 9) + (lane << 4));
            acc.v = __builtin_amdgcn_wmma_f32_16x16x32_f16(
                false, af.v, false, bf, (short)0, acc.v, false, false);
        }
    }
    #pragma unroll
    for (int i = 0; i < 8; i++) {
        int m = row0 + i + 8 * hi;
        if (m < M) C[(long long)m * KCH + col] = acc.f[i];
    }
}

// ---------------------------------------------------------------------------
// Edge forward 1: vec, r, unit, radial; scatter radial@Wbf into m1. Wave/edge.
// ---------------------------------------------------------------------------
__global__ __launch_bounds__(256) void edge_fwd1(
    const float* __restrict__ pos, const float* __restrict__ shifts,
    const int* __restrict__ ei, const float* __restrict__ Wbf,
    float* __restrict__ e_radial, float* __restrict__ e_unit,
    float* __restrict__ e_r, float* __restrict__ m1)
{
    int e = (blockIdx.x * 256 + threadIdx.x) >> 5;
    int l = threadIdx.x & 31;
    if (e >= NEDGE) return;
    int s = ei[e], t = ei[NEDGE + e];
    float v0 = pos[t * 3 + 0] - pos[s * 3 + 0] + shifts[e * 3 + 0];
    float v1 = pos[t * 3 + 1] - pos[s * 3 + 1] + shifts[e * 3 + 1];
    float v2 = pos[t * 3 + 2] - pos[s * 3 + 2] + shifts[e * 3 + 2];
    float r = sqrtf(v0 * v0 + v1 * v1 + v2 * v2 + 1e-12f);
    float inv = 1.f / r;
    if (l == 0) {
        e_r[e] = r;
        e_unit[e * 3 + 0] = v0 * inv;
        e_unit[e * 3 + 1] = v1 * inv;
        e_unit[e * 3 + 2] = v2 * inv;
    }
    float env = (r < 5.f) ? 0.5f * (__cosf(3.14159265358979f * r * 0.2f) + 1.f) : 0.f;
    float rad[8];
    #pragma unroll
    for (int j = 0; j < 8; j++) {
        float d = r - (float)j * (5.f / 7.f);
        rad[j] = __expf(-2.56f * d * d) * env;
    }
    if (l < 8) e_radial[e * 8 + l] = rad[l];
    #pragma unroll
    for (int u = 0; u < 4; u++) {
        int k = l + 32 * u;
        float a = 0.f;
        #pragma unroll
        for (int j = 0; j < 8; j++) a += rad[j] * Wbf[j * KCH + k];
        atomicAdd(&m1[(long long)t * KCH + k], a);
    }
}

// ---------------------------------------------------------------------------
// Edge forward 2: radial MLP -> w0,w1; A0e/A1e scatter to S0,S1. Wave/edge.
// ---------------------------------------------------------------------------
__global__ __launch_bounds__(256) void edge_fwd2(
    const float* __restrict__ e_radial, const float* __restrict__ e_unit,
    const float* __restrict__ h1p, const int* __restrict__ ei,
    const float* __restrict__ Wr1, const float* __restrict__ br1,
    const float* __restrict__ Wr2, const float* __restrict__ br2,
    const float* __restrict__ Wr3, const float* __restrict__ br3,
    const float* __restrict__ Wr4, const float* __restrict__ br4,
    float* __restrict__ S0, float* __restrict__ S1)
{
    int e = (blockIdx.x * 256 + threadIdx.x) >> 5;
    int l = threadIdx.x & 31;
    if (e >= NEDGE) return;
    int s = ei[e], t = ei[NEDGE + e];
    __builtin_prefetch(h1p + (long long)s * KCH, 0, 1);   // global_prefetch_b8
    float rad[8];
    #pragma unroll
    for (int j = 0; j < 8; j++) rad[j] = e_radial[e * 8 + j];

    float pre1 = br1[l];
    #pragma unroll
    for (int j = 0; j < 8; j++) pre1 += rad[j] * Wr1[j * 32 + l];
    float ha = siluf(pre1);
    float pre2 = br2[l];
    for (int j = 0; j < 32; j++) pre2 += __shfl(ha, j, 32) * Wr2[j * 32 + l];
    float hb = siluf(pre2);
    float pre3 = br3[l];
    for (int j = 0; j < 32; j++) pre3 += __shfl(hb, j, 32) * Wr3[j * 32 + l];
    float hc = siluf(pre3);

    float rw[8];
    #pragma unroll
    for (int u = 0; u < 8; u++) rw[u] = br4[l + 32 * u];
    for (int j = 0; j < 32; j++) {
        float h = __shfl(hc, j, 32);
        const float* wr = Wr4 + j * 256 + l;
        #pragma unroll
        for (int u = 0; u < 8; u++) rw[u] += h * wr[32 * u];
    }
    float un0 = e_unit[e * 3 + 0], un1 = e_unit[e * 3 + 1], un2 = e_unit[e * 3 + 2];
    #pragma unroll
    for (int u = 0; u < 4; u++) {
        int k = l + 32 * u;
        float hs  = h1p[(long long)s * KCH + k];
        float a0e = rw[u] * hs;          // w0 * hs
        float wh  = rw[4 + u] * hs;      // w1 * hs
        atomicAdd(&S0[(long long)t * KCH + k], a0e);
        atomicAdd(&S1[((long long)t * 3 + 0) * KCH + k], un0 * wh);
        atomicAdd(&S1[((long long)t * 3 + 1) * KCH + k], un1 * wh);
        atomicAdd(&S1[((long long)t * 3 + 2) * KCH + k], un2 * wh);
    }
}

// ---------------------------------------------------------------------------
// B2 = c1*A0 + c2a*A0^2 + c2b*sum_m A1m^2
// ---------------------------------------------------------------------------
__global__ __launch_bounds__(256) void node_B2_k(
    const int* __restrict__ spec, const float* __restrict__ A0,
    const float* __restrict__ A1, const float* __restrict__ Wc1,
    const float* __restrict__ Wc2a, const float* __restrict__ Wc2b,
    float* __restrict__ B2)
{
    int idx = blockIdx.x * 256 + threadIdx.x;
    if (idx >= NNODE * KCH) return;
    int n = idx >> 7, k = idx & 127;
    int sp = spec[n];
    float a0 = A0[idx];
    float q = 0.f;
    #pragma unroll
    for (int m = 0; m < 3; m++) {
        float a = A1[((long long)n * 3 + m) * KCH + k];
        q += a * a;
    }
    B2[idx] = Wc1[sp * KCH + k] * a0 + Wc2a[sp * KCH + k] * a0 * a0 + Wc2b[sp * KCH + k] * q;
}

// ---------------------------------------------------------------------------
// Readout: E0+E1+E2 -> atomic E_batch; seed g_h2. Wave/node.
// ---------------------------------------------------------------------------
__global__ __launch_bounds__(256) void readout_k(
    const float* __restrict__ h1, const float* __restrict__ h2,
    const int* __restrict__ spec, const int* __restrict__ batch,
    const float* __restrict__ Wr0, const float* __restrict__ br0,
    const float* __restrict__ Wro1, const float* __restrict__ bro1,
    const float* __restrict__ Wro2a, const float* __restrict__ bro2a,
    const float* __restrict__ Wro2b, const float* __restrict__ bro2b,
    float* __restrict__ Ebatch, float* __restrict__ gh2)
{
    int n = (blockIdx.x * 256 + threadIdx.x) >> 5;
    int l = threadIdx.x & 31;
    if (n >= NNODE) return;

    float e1 = 0.f;
    #pragma unroll
    for (int u = 0; u < 4; u++) {
        int k = l + 32 * u;
        e1 += h1[(long long)n * KCH + k] * Wro1[k];
    }
    for (int off = 16; off; off >>= 1) e1 += __shfl_xor(e1, off, 32);

    int o = l & 15;
    int kh = (l >> 4) * 64;
    float pre = 0.f;
    for (int kk = 0; kk < 64; kk++) {
        int k = kh + kk;
        pre += h2[(long long)n * KCH + k] * Wro2a[k * 16 + o];
    }
    pre += __shfl_xor(pre, 16, 32);
    pre += bro2a[o];
    float sg = 1.f / (1.f + __expf(-pre));
    float tv = pre * sg;
    float dv = sg * (1.f + pre * (1.f - sg));
    float w2b = Wro2b[o];
    float tc = tv * w2b;
    float dsc = dv * w2b;

    float e2 = tc;
    for (int off = 16; off; off >>= 1) e2 += __shfl_xor(e2, off, 32);
    e2 *= 0.5f;

    float gh[4] = {0.f, 0.f, 0.f, 0.f};
    for (int o2 = 0; o2 < 16; o2++) {
        float d = __shfl(dsc, o2, 32);
        #pragma unroll
        for (int u = 0; u < 4; u++) {
            int k = l + 32 * u;
            gh[u] += Wro2a[k * 16 + o2] * d;
        }
    }
    #pragma unroll
    for (int u = 0; u < 4; u++) gh2[(long long)n * KCH + l + 32 * u] = gh[u];

    if (l == 0) {
        float en = Wr0[spec[n]] + br0[0] + e1 + bro1[0] + e2 + bro2b[0];
        atomicAdd(&Ebatch[batch[n]], en);
    }
}

// ---------------------------------------------------------------------------
// g_A0 = g_B2*(c1 + 2 c2a A0) (in place); g_A1m = 2 c2b g_B2 A1m (in place)
// ---------------------------------------------------------------------------
__global__ __launch_bounds__(256) void node_gA_k(
    const int* __restrict__ spec, const float* __restrict__ gB2,
    float* __restrict__ A0, float* __restrict__ A1,
    const float* __restrict__ Wc1, const float* __restrict__ Wc2a,
    const float* __restrict__ Wc2b)
{
    int idx = blockIdx.x * 256 + threadIdx.x;
    if (idx >= NNODE * KCH) return;
    int n = idx >> 7, k = idx & 127;
    int sp = spec[n];
    float g = gB2[idx];
    float a0 = A0[idx];
    A0[idx] = g * (Wc1[sp * KCH + k] + 2.f * Wc2a[sp * KCH + k] * a0);
    float c2b2 = 2.f * Wc2b[sp * KCH + k] * g;
    #pragma unroll
    for (int m = 0; m < 3; m++) {
        long long i = ((long long)n * 3 + m) * KCH + k;
        A1[i] = c2b2 * A1[i];
    }
}

// ---------------------------------------------------------------------------
// Edge backward A: recompute MLP; g_w -> MLP backward -> e_gradb; g_unit ->
// e_gunit; scatter g_hs into g_h1p[sender]. Wave/edge.
// ---------------------------------------------------------------------------
__global__ __launch_bounds__(256) void edge_bwdA(
    const float* __restrict__ e_radial, const float* __restrict__ e_unit,
    const float* __restrict__ h1p, const float* __restrict__ gS0,
    const float* __restrict__ gS1, const int* __restrict__ ei,
    const float* __restrict__ Wr1, const float* __restrict__ br1,
    const float* __restrict__ Wr2, const float* __restrict__ br2,
    const float* __restrict__ Wr3, const float* __restrict__ br3,
    const float* __restrict__ Wr4, const float* __restrict__ br4,
    float* __restrict__ gh1p, float* __restrict__ e_gradb,
    float* __restrict__ e_gunit)
{
    int e = (blockIdx.x * 256 + threadIdx.x) >> 5;
    int l = threadIdx.x & 31;
    if (e >= NEDGE) return;
    int s = ei[e], t = ei[NEDGE + e];
    __builtin_prefetch(h1p + (long long)s * KCH, 0, 1);
    __builtin_prefetch(gS0 + (long long)t * KCH, 0, 1);
    __builtin_prefetch(gS1 + (long long)t * 3 * KCH, 0, 1);

    float rad[8];
    #pragma unroll
    for (int j = 0; j < 8; j++) rad[j] = e_radial[e * 8 + j];
    float pre1 = br1[l];
    #pragma unroll
    for (int j = 0; j < 8; j++) pre1 += rad[j] * Wr1[j * 32 + l];
    float ha = siluf(pre1);
    float pre2 = br2[l];
    for (int j = 0; j < 32; j++) pre2 += __shfl(ha, j, 32) * Wr2[j * 32 + l];
    float hb = siluf(pre2);
    float pre3 = br3[l];
    for (int j = 0; j < 32; j++) pre3 += __shfl(hb, j, 32) * Wr3[j * 32 + l];
    float hc = siluf(pre3);
    float rw[8];
    #pragma unroll
    for (int u = 0; u < 8; u++) rw[u] = br4[l + 32 * u];
    for (int j = 0; j < 32; j++) {
        float h = __shfl(hc, j, 32);
        const float* wr = Wr4 + j * 256 + l;
        #pragma unroll
        for (int u = 0; u < 8; u++) rw[u] += h * wr[32 * u];
    }

    float un0 = e_unit[e * 3 + 0], un1 = e_unit[e * 3 + 1], un2 = e_unit[e * 3 + 2];
    float gRw[8];
    float gu0 = 0.f, gu1 = 0.f, gu2 = 0.f;
    #pragma unroll
    for (int u = 0; u < 4; u++) {
        int k = l + 32 * u;
        float hs  = h1p[(long long)s * KCH + k];
        float g0  = gS0[(long long)t * KCH + k];
        float g10 = gS1[((long long)t * 3 + 0) * KCH + k];
        float g11 = gS1[((long long)t * 3 + 1) * KCH + k];
        float g12 = gS1[((long long)t * 3 + 2) * KCH + k];
        float w0v = rw[u], w1v = rw[4 + u];
        float wh = w1v * hs;
        gu0 += g10 * wh; gu1 += g11 * wh; gu2 += g12 * wh;
        float gwh = g10 * un0 + g11 * un1 + g12 * un2;
        float ghs = g0 * w0v + gwh * w1v;
        atomicAdd(&gh1p[(long long)s * KCH + k], ghs);
        gRw[u]     = g0 * hs;
        gRw[4 + u] = gwh * hs;
    }
    for (int off = 16; off; off >>= 1) {
        gu0 += __shfl_xor(gu0, off, 32);
        gu1 += __shfl_xor(gu1, off, 32);
        gu2 += __shfl_xor(gu2, off, 32);
    }
    if (l == 0) {
        e_gunit[e * 3 + 0] = gu0;
        e_gunit[e * 3 + 1] = gu1;
        e_gunit[e * 3 + 2] = gu2;
    }

    float ghc = 0.f;
    for (int sl = 0; sl < 32; sl++) {
        const float* wr = Wr4 + l * 256 + sl;
        #pragma unroll
        for (int u = 0; u < 8; u++) {
            float gv = __shfl(gRw[u], sl, 32);
            ghc += gv * wr[32 * u];
        }
    }
    float gp3 = ghc * dsiluf(pre3);
    float ghb = 0.f;
    for (int i = 0; i < 32; i++) ghb += __shfl(gp3, i, 32) * Wr3[l * 32 + i];
    float gp2 = ghb * dsiluf(pre2);
    float gha = 0.f;
    for (int i = 0; i < 32; i++) gha += __shfl(gp2, i, 32) * Wr2[l * 32 + i];
    float gp1 = gha * dsiluf(pre1);
    float grb = 0.f;
    int jr = l & 7;
    for (int i = 0; i < 32; i++) grb += __shfl(gp1, i, 32) * Wr1[jr * 32 + i];
    if (l < 8) e_gradb[e * 8 + l] = grb;
}

// ---------------------------------------------------------------------------
// Edge backward B: g_radial (m1 path + stored MLP path) -> g_r; + g_unit ->
// g_vec; scatter forces. Wave/edge.
// ---------------------------------------------------------------------------
__global__ __launch_bounds__(256) void edge_bwdB(
    const float* __restrict__ e_r, const float* __restrict__ e_unit,
    const float* __restrict__ e_gunit, const float* __restrict__ e_gradb,
    const float* __restrict__ gm1, const float* __restrict__ Wbf,
    const int* __restrict__ ei, float* __restrict__ F)
{
    int e = (blockIdx.x * 256 + threadIdx.x) >> 5;
    int l = threadIdx.x & 31;
    if (e >= NEDGE) return;
    int s = ei[e], t = ei[NEDGE + e];
    __builtin_prefetch(gm1 + (long long)t * KCH, 0, 1);

    float part[8] = {0, 0, 0, 0, 0, 0, 0, 0};
    #pragma unroll
    for (int u = 0; u < 4; u++) {
        int k = l + 32 * u;
        float g = gm1[(long long)t * KCH + k];
        #pragma unroll
        for (int j = 0; j < 8; j++) part[j] += g * Wbf[j * KCH + k];
    }
    #pragma unroll
    for (int j = 0; j < 8; j++)
        for (int off = 16; off; off >>= 1) part[j] += __shfl_xor(part[j], off, 32);

    float r = e_r[e];
    float env = 0.f, denv = 0.f;
    if (r < 5.f) {
        env  = 0.5f * (__cosf(3.14159265358979f * r * 0.2f) + 1.f);
        denv = -0.5f * 3.14159265358979f * 0.2f * __sinf(3.14159265358979f * r * 0.2f);
    }
    float gr = 0.f;
    #pragma unroll
    for (int j = 0; j < 8; j++) {
        float d = r - (float)j * (5.f / 7.f);
        float rbf = __expf(-2.56f * d * d);
        float drbf = -2.f * 2.56f * d * rbf;
        float gtot = part[j] + e_gradb[e * 8 + j];
        gr += gtot * (drbf * env + rbf * denv);
    }
    float u0 = e_unit[e * 3 + 0], u1 = e_unit[e * 3 + 1], u2 = e_unit[e * 3 + 2];
    float g0 = e_gunit[e * 3 + 0], g1 = e_gunit[e * 3 + 1], g2 = e_gunit[e * 3 + 2];
    float gdot = g0 * u0 + g1 * u1 + g2 * u2;
    float inv = 1.f / r;
    float gv0 = gr * u0 + (g0 - gdot * u0) * inv;
    float gv1 = gr * u1 + (g1 - gdot * u1) * inv;
    float gv2 = gr * u2 + (g2 - gdot * u2) * inv;
    if (l == 0) {
        atomicAdd(&F[t * 3 + 0], -gv0);
        atomicAdd(&F[t * 3 + 1], -gv1);
        atomicAdd(&F[t * 3 + 2], -gv2);
        atomicAdd(&F[s * 3 + 0], gv0);
        atomicAdd(&F[s * 3 + 1], gv1);
        atomicAdd(&F[s * 3 + 2], gv2);
    }
}

// ---------------------------------------------------------------------------
extern "C" void kernel_launch(void* const* d_in, const int* in_sizes, int n_in,
                              void* d_out, int out_size, void* d_ws, size_t ws_size,
                              hipStream_t stream)
{
    (void)in_sizes; (void)n_in; (void)ws_size;
    const float* pos    = (const float*)d_in[0];
    const float* shifts = (const float*)d_in[1];
    const int*   ei     = (const int*)d_in[2];
    const int*   spec   = (const int*)d_in[3];
    const int*   batch  = (const int*)d_in[4];
    const float* Wh0    = (const float*)d_in[5];
    const float* bh0    = (const float*)d_in[6];
    const float* Wbf    = (const float*)d_in[7];
    const float* W_h0h1 = (const float*)d_in[8];
    const float* W_m1h1 = (const float*)d_in[9];
    const float* W_h1h1p= (const float*)d_in[10];
    const float* Wr1    = (const float*)d_in[11];
    const float* br1    = (const float*)d_in[12];
    const float* Wr2    = (const float*)d_in[13];
    const float* br2    = (const float*)d_in[14];
    const float* Wr3    = (const float*)d_in[15];
    const float* br3    = (const float*)d_in[16];
    const float* Wr4    = (const float*)d_in[17];
    const float* br4    = (const float*)d_in[18];
    const float* WA0    = (const float*)d_in[19];
    const float* WA1    = (const float*)d_in[20];
    const float* Wc1    = (const float*)d_in[21];
    const float* Wc2a   = (const float*)d_in[22];
    const float* Wc2b   = (const float*)d_in[23];
    const float* W_h1h2 = (const float*)d_in[24];
    const float* W_m2h2 = (const float*)d_in[25];
    const float* Wr0    = (const float*)d_in[26];
    const float* br0    = (const float*)d_in[27];
    const float* Wro1   = (const float*)d_in[28];
    const float* bro1   = (const float*)d_in[29];
    const float* Wro2a  = (const float*)d_in[30];
    const float* bro2a  = (const float*)d_in[31];
    const float* Wro2b  = (const float*)d_in[32];
    const float* bro2b  = (const float*)d_in[33];

    float* out = (float*)d_out;
    float* wsf = (float*)d_ws;
    const long long EE = NEDGE, S = (long long)NNODE * KCH;
    float* e_radial = wsf;
    float* e_unit   = wsf + 8 * EE;
    float* e_r      = wsf + 11 * EE;
    float* e_gradb  = wsf + 12 * EE;
    float* e_gunit  = wsf + 20 * EE;
    float* nb       = wsf + 23 * EE;
    float* n_h0   = nb + 0 * S;
    float* n_m1   = nb + 1 * S;
    float* n_h1   = nb + 2 * S;
    float* n_h1p  = nb + 3 * S;
    float* n_S0   = nb + 4 * S;   // later g_S0
    float* n_S1   = nb + 5 * S;   // 3 slots; later g_S1
    float* n_A0   = nb + 8 * S;   // later g_A0 (in place)
    float* n_A1   = nb + 9 * S;   // 3 slots; later g_A1 (in place)
    float* n_B2   = nb + 12 * S;
    float* n_h2   = nb + 13 * S;
    float* n_gh2  = nb + 14 * S;
    float* n_gB2  = nb + 15 * S;
    float* n_gh1p = nb + 16 * S;
    float* n_gh1  = nb + 17 * S;
    float* n_gm1  = nb + 18 * S;
    _Float16* wh  = (_Float16*)(nb + 19 * S);  // 13 x 128x128 f16 (swizzled)

    const int NKB = (NNODE * KCH) / 256;
    const int EB  = NEDGE / 8;
    const int NB  = NNODE / 8;
    const int GN  = (NNODE + 15) / 16;
    const int G3N = (3 * NNODE + 15) / 16;

    // ---- forward ----
    prep_weights<<<(13 * 16384 + 255) / 256, 256, 0, stream>>>(
        W_h0h1, W_m1h1, W_h1h1p, WA0, WA1, W_h1h2, W_m2h2, wh);
    node_h0_k<<<NKB, 256, 0, stream>>>(Wh0, bh0, spec, n_h0);
    hipMemsetAsync(n_m1, 0, S * sizeof(float), stream);
    edge_fwd1<<<EB, 256, 0, stream>>>(pos, shifts, ei, Wbf, e_radial, e_unit, e_r, n_m1);
    gemm_nk<<<GN, 256, 0, stream>>>(n_h0, wh + 0 * 16384, n_m1, wh + 1 * 16384,
                                    nullptr, n_h1, NNODE, 0);
    gemm_nk<<<GN, 256, 0, stream>>>(n_h1, wh + 2 * 16384, nullptr, nullptr,
                                    nullptr, n_h1p, NNODE, 0);
    hipMemsetAsync(n_S0, 0, 4 * S * sizeof(float), stream);  // S0 + S1
    edge_fwd2<<<EB, 256, 0, stream>>>(e_radial, e_unit, n_h1p, ei,
                                      Wr1, br1, Wr2, br2, Wr3, br3, Wr4, br4, n_S0, n_S1);
    gemm_nk<<<GN, 256, 0, stream>>>(n_S0, wh + 3 * 16384, nullptr, nullptr,
                                    nullptr, n_A0, NNODE, 0);
    gemm_nk<<<G3N, 256, 0, stream>>>(n_S1, wh + 4 * 16384, nullptr, nullptr,
                                     nullptr, n_A1, 3 * NNODE, 0);
    node_B2_k<<<NKB, 256, 0, stream>>>(spec, n_A0, n_A1, Wc1, Wc2a, Wc2b, n_B2);
    gemm_nk<<<GN, 256, 0, stream>>>(n_h1, wh + 5 * 16384, n_B2, wh + 6 * 16384,
                                    nullptr, n_h2, NNODE, 0);
    hipMemsetAsync(out, 0, (size_t)out_size * sizeof(float), stream);
    readout_k<<<NB, 256, 0, stream>>>(n_h1, n_h2, spec, batch,
                                      Wr0, br0, Wro1, bro1, Wro2a, bro2a, Wro2b, bro2b,
                                      out, n_gh2);
    // ---- backward ----
    gemm_nk<<<GN, 256, 0, stream>>>(n_gh2, wh + 11 * 16384, nullptr, nullptr,
                                    nullptr, n_gB2, NNODE, 0);            // g_B2
    node_gA_k<<<NKB, 256, 0, stream>>>(spec, n_gB2, n_A0, n_A1, Wc1, Wc2a, Wc2b);
    gemm_nk<<<GN, 256, 0, stream>>>(n_A0, wh + 8 * 16384, nullptr, nullptr,
                                    nullptr, n_S0, NNODE, 0);             // g_S0
    gemm_nk<<<G3N, 256, 0, stream>>>(n_A1, wh + 9 * 16384, nullptr, nullptr,
                                     nullptr, n_S1, 3 * NNODE, 0);        // g_S1
    gemm_nk<<<GN, 256, 0, stream>>>(n_gh2, wh + 10 * 16384, nullptr, nullptr,
                                    Wro1, n_gh1, NNODE, 0);               // g_h1 partial
    hipMemsetAsync(n_gh1p, 0, S * sizeof(float), stream);
    edge_bwdA<<<EB, 256, 0, stream>>>(e_radial, e_unit, n_h1p, n_S0, n_S1, ei,
                                      Wr1, br1, Wr2, br2, Wr3, br3, Wr4, br4,
                                      n_gh1p, e_gradb, e_gunit);
    gemm_nk<<<GN, 256, 0, stream>>>(n_gh1p, wh + 7 * 16384, nullptr, nullptr,
                                    nullptr, n_gh1, NNODE, 1);            // g_h1 += g_h1p@W^T
    gemm_nk<<<GN, 256, 0, stream>>>(n_gh1, wh + 12 * 16384, nullptr, nullptr,
                                    nullptr, n_gm1, NNODE, 0);            // g_m1
    edge_bwdB<<<EB, 256, 0, stream>>>(e_r, e_unit, e_gunit, e_gradb, n_gm1, Wbf,
                                      ei, out + NGR);
}
